// GNN_46918222742125
// MI455X (gfx1250) — compile-verified, matrix operand order checked by reading, not run on hardware
//
#include <hip/hip_runtime.h>

typedef unsigned short u16;
typedef __attribute__((ext_vector_type(16))) __bf16 v16bf;
typedef __attribute__((ext_vector_type(8)))  float  v8f;

__device__ __forceinline__ u16 f2bf(float f) {
  unsigned u = __float_as_uint(f);
  unsigned r = u + 0x7FFFu + ((u >> 16) & 1u);   // round-to-nearest-even
  return (u16)(r >> 16);
}
__device__ __forceinline__ __bf16 bfbits(u16 v) { return __builtin_bit_cast(__bf16, v); }

// ---------------------------------------------------------------------------
// f32 -> bf16 convert (grid-stride), row-major copy (for A-side matrices)
// ---------------------------------------------------------------------------
__global__ void gnn_cvt_bf16(const float* __restrict__ in, u16* __restrict__ out,
                             unsigned long long n) {
  unsigned long long i = (unsigned long long)blockIdx.x * blockDim.x + threadIdx.x;
  unsigned long long stride = (unsigned long long)gridDim.x * blockDim.x;
  for (; i < n; i += stride) out[i] = f2bf(in[i]);
}

// ---------------------------------------------------------------------------
// f32 -> bf16 transpose-convert: src[R,C] f32 -> dst[C,ldDst] bf16 (dst[c][r]),
// zero-filled for r in [R, ldDst). Tiled through LDS for coalescing.
// ---------------------------------------------------------------------------
__global__ __launch_bounds__(256) void gnn_cvtT(const float* __restrict__ src,
                                                u16* __restrict__ dst,
                                                int R, int C, int ldDst) {
  __shared__ u16 tile[32][33];
  int c0 = blockIdx.x * 32;  // src col == dst row
  int r0 = blockIdx.y * 32;  // src row == dst col
  for (int i = threadIdx.y; i < 32; i += 8) {
    int r = r0 + i, c = c0 + threadIdx.x;
    float v = (r < R && c < C) ? src[(size_t)r * C + c] : 0.f;
    tile[threadIdx.x][i] = f2bf(v);            // tile[cLocal][rLocal]
  }
  __syncthreads();
  for (int i = threadIdx.y; i < 32; i += 8) {
    int dr = c0 + i;                           // dst row (src col)
    int dc = r0 + threadIdx.x;                 // dst col (src row)
    if (dr < C && dc < ldDst) dst[(size_t)dr * ldDst + dc] = tile[i][threadIdx.x];
  }
}

// ---------------------------------------------------------------------------
// bf16 WMMA GEMM: C[M,N] = A[M,K] @ Bt[N,K]^T (+bias[N]), fp32 accumulate.
// Block tile 128x128, BK=32, 8 waves; wave tile 64x32 = 4x2 wmma fragments.
// A and Bt are both row-major with contiguous K (Bt pre-transposed), so tile
// loads are symmetric 16-byte vector loads on interior tiles.
// ---------------------------------------------------------------------------
#define GT_BM 128
#define GT_BN 128
#define GT_BK 32
#define GT_LDT 40   // padded LDS row (u16); 80B rows keep 16B alignment

__global__ __launch_bounds__(256) void gnn_gemm_bf16(
    const u16* __restrict__ A, const u16* __restrict__ Bt,
    const float* __restrict__ bias, float* __restrict__ C,
    int M, int N, int K) {
  __shared__ u16 As[GT_BM * GT_LDT];
  __shared__ u16 Bs[GT_BN * GT_LDT];

  const int tid  = threadIdx.x;
  const int wave = tid >> 5;
  const int lane = tid & 31;
  const int hf   = lane >> 4;
  const int ml   = lane & 15;
  const int wm   = wave & 1;     // 2 wave-rows of 64
  const int wn   = wave >> 1;    // 4 wave-cols of 32
  const int bm0  = blockIdx.y * GT_BM;
  const int bn0  = blockIdx.x * GT_BN;
  const bool fullMN = (bm0 + GT_BM <= M) && (bn0 + GT_BN <= N);

  v8f acc[4][2];
#pragma unroll
  for (int i = 0; i < 4; ++i)
#pragma unroll
    for (int j = 0; j < 2; ++j) acc[i][j] = v8f{};

  const int ktiles = (K + GT_BK - 1) / GT_BK;
  for (int kt = 0; kt < ktiles; ++kt) {
    const int k0 = kt * GT_BK;
    if (fullMN && (k0 + GT_BK <= K)) {
      // fast path: 512 16-byte chunks per matrix, 2 per thread
#pragma unroll
      for (int i = 0; i < 2; ++i) {
        int cid = i * 256 + tid;                 // 0..511
        int r = cid >> 2, cp = (cid & 3) << 3;   // row, col-offset (8 u16)
        *(uint4*)(As + r * GT_LDT + cp) =
            *(const uint4*)(A + (size_t)(bm0 + r) * K + k0 + cp);
        *(uint4*)(Bs + r * GT_LDT + cp) =
            *(const uint4*)(Bt + (size_t)(bn0 + r) * K + k0 + cp);
      }
      if (k0 + 2 * GT_BK <= K) {                 // prefetch next k-tile
        int r = tid >> 2, cp = (tid & 3) << 3;
        __builtin_prefetch(A + (size_t)(bm0 + r) * K + k0 + GT_BK + cp, 0, 1);
        __builtin_prefetch(Bt + (size_t)(bn0 + r) * K + k0 + GT_BK + cp, 0, 1);
      }
    } else {
      // edge path: guarded scalar loads (only edge blocks / k-tails)
#pragma unroll
      for (int i = 0; i < 16; ++i) {
        int idx = i * 256 + tid;
        int r = idx >> 5, c = idx & 31;
        int gc = k0 + c;
        u16 va = 0, vb = 0;
        if (bm0 + r < M && gc < K) va = A[(size_t)(bm0 + r) * K + gc];
        if (bn0 + r < N && gc < K) vb = Bt[(size_t)(bn0 + r) * K + gc];
        As[r * GT_LDT + c] = va;
        Bs[r * GT_LDT + c] = vb;
      }
    }
    __syncthreads();

    // ISA 16-bit 16x32 layout: VGPR v (pair e=2v,2v+1):
    //   k = (v>=4)*16 + (v%4)*2 + half*8 (+0/+1)
    v16bf afr[4], bfr[2];
#pragma unroll
    for (int mf = 0; mf < 4; ++mf) {
      const u16* ap = &As[(wm * 64 + mf * 16 + ml) * GT_LDT];
#pragma unroll
      for (int v = 0; v < 8; ++v) {
        int kk = ((v & 4) << 2) + ((v & 3) << 1) + (hf << 3);
        afr[mf][2 * v]     = bfbits(ap[kk]);
        afr[mf][2 * v + 1] = bfbits(ap[kk + 1]);
      }
    }
#pragma unroll
    for (int nf = 0; nf < 2; ++nf) {
      const u16* bp = &Bs[(wn * 32 + nf * 16 + ml) * GT_LDT];
#pragma unroll
      for (int v = 0; v < 8; ++v) {
        int kk = ((v & 4) << 2) + ((v & 3) << 1) + (hf << 3);
        bfr[nf][2 * v]     = bfbits(bp[kk]);
        bfr[nf][2 * v + 1] = bfbits(bp[kk + 1]);
      }
    }
#pragma unroll
    for (int mf = 0; mf < 4; ++mf)
#pragma unroll
      for (int nf = 0; nf < 2; ++nf)
        acc[mf][nf] = __builtin_amdgcn_wmma_f32_16x16x32_bf16(
            false, afr[mf], false, bfr[nf], (short)0, acc[mf][nf], false, false);
    __syncthreads();
  }

  // C fragment: VGPR r, lane: m = r + half*8, n = lane%16
#pragma unroll
  for (int mf = 0; mf < 4; ++mf) {
#pragma unroll
    for (int nf = 0; nf < 2; ++nf) {
      int gn = bn0 + wn * 32 + nf * 16 + ml;
#pragma unroll
      for (int r = 0; r < 8; ++r) {
        int gm = bm0 + wm * 64 + mf * 16 + hf * 8 + r;
        if (gm < M && gn < N) {
          float v = acc[mf][nf][r];
          if (bias) v += bias[gn];
          C[(size_t)gm * N + gn] = v;
        }
      }
    }
  }
}

// ---------------------------------------------------------------------------
// Column mean / biased var over M rows of X[M,N]
// ---------------------------------------------------------------------------
__global__ void gnn_colstats(const float* __restrict__ X, int M, int N,
                             float* __restrict__ mean, float* __restrict__ var) {
  __shared__ float sS[4][64];
  __shared__ float sQ[4][64];
  int tx = threadIdx.x, ty = threadIdx.y;
  int c = blockIdx.x * 64 + tx;
  float s = 0.f, q = 0.f;
  if (c < N)
    for (int r = ty; r < M; r += 4) {
      float v = X[(size_t)r * N + c];
      s += v; q += v * v;
    }
  sS[ty][tx] = s; sQ[ty][tx] = q;
  __syncthreads();
  if (ty == 0 && c < N) {
    float S = sS[0][tx] + sS[1][tx] + sS[2][tx] + sS[3][tx];
    float Q = sQ[0][tx] + sQ[1][tx] + sQ[2][tx] + sQ[3][tx];
    float m = S / (float)M;
    mean[c] = m;
    var[c]  = Q / (float)M - m * m;
  }
}

// ---------------------------------------------------------------------------
// Fused BatchNorm(training) + LeakyReLU(0.01); optional f32 and bf16 outputs
// ---------------------------------------------------------------------------
__global__ void gnn_bn_lrelu(const float* __restrict__ X, int M, int N,
                             const float* __restrict__ g, const float* __restrict__ b,
                             const float* __restrict__ mean, const float* __restrict__ var,
                             float* __restrict__ outF, u16* __restrict__ outBF) {
  size_t total = (size_t)M * N;
  size_t stride = (size_t)gridDim.x * blockDim.x;
  for (size_t i = (size_t)blockIdx.x * blockDim.x + threadIdx.x; i < total; i += stride) {
    int c = (int)(i % (size_t)N);
    float v = X[i];
    float z = g[c] * (v - mean[c]) * rsqrtf(var[c] + 1e-5f) + b[c];
    z = (z >= 0.f) ? z : 0.01f * z;
    if (outF)  outF[i]  = z;
    if (outBF) outBF[i] = f2bf(z);
  }
}

// ---------------------------------------------------------------------------
// Conv branch via 26-symbol bucketing:
//   C[b,o,s,k] = sum_{i: pe[b,i]==s} W_conv[o,i,k]   (LDS accumulate)
//   conv[b,o,h] = b_conv[o] + sum_{s,k} C[b,o,s,k] * emb[s,h+k]
// Output bf16 rows [b, o*121+h] for the FC GEMM (A-side, row stride 3872).
// ---------------------------------------------------------------------------
__global__ __launch_bounds__(256) void gnn_conv(
    const int* __restrict__ pe, const float* __restrict__ Wc,
    const float* __restrict__ emb, const float* __restrict__ bconv,
    u16* __restrict__ out) {
  __shared__ float accS[32 * 209];   // [o][s*8+k], stride 209 avoids bank clustering
  __shared__ float embS[26 * 128];
  __shared__ int   eS[1000];
  const int b = blockIdx.x;
  const int t = threadIdx.x;
  for (int i = t; i < 32 * 209; i += 256) accS[i] = 0.f;
  for (int i = t; i < 26 * 128; i += 256) embS[i] = emb[i];
  for (int i = t; i < 1000; i += 256)     eS[i]   = pe[(size_t)b * 1000 + i];
  __syncthreads();

  const int o = t >> 3, k = t & 7;     // 32x8 = 256 exact
  float* arow = &accS[o * 209 + k];
  const float* wrow = &Wc[(size_t)o * 8000 + k];
  for (int i = 0; i < 1000; ++i) {
    int s = eS[i];
    arow[s * 8] += wrow[i * 8];        // each thread owns its (o,k) slots
  }
  __syncthreads();

  for (int idx = t; idx < 3872; idx += 256) {
    int oo = idx / 121, h = idx - oo * 121;
    float sum = bconv[oo];
    const float* ac = &accS[oo * 209];
#pragma unroll
    for (int s = 0; s < 26; ++s) {
      const float* eb = &embS[s * 128 + h];
#pragma unroll
      for (int kk = 0; kk < 8; ++kk) sum += ac[s * 8 + kk] * eb[kk];
    }
    out[(size_t)b * 3872 + idx] = f2bf(sum);
  }
}

// ---------------------------------------------------------------------------
// feature = concat(d_vecs[300], ecfps[d_index][1024], xt[128], gos[p_index][1024])
// f32 (stride 2476) into d_out; bf16 mirror padded to stride 2496 (zeros).
// ---------------------------------------------------------------------------
__global__ void gnn_feature(const float* __restrict__ dvecs, const float* __restrict__ H1,
                            const float* __restrict__ xt, const float* __restrict__ H2,
                            const int* __restrict__ dind, const int* __restrict__ pind,
                            float* __restrict__ featF, u16* __restrict__ featB) {
  const int b = blockIdx.x;
  const int di = dind[b], pi = pind[b];
  for (int c = threadIdx.x; c < 2496; c += blockDim.x) {
    float v = 0.f;
    if (c < 300)       v = dvecs[(size_t)b * 300 + c];
    else if (c < 1324) v = H1[(size_t)di * 1024 + (c - 300)];
    else if (c < 1452) v = xt[(size_t)b * 128 + (c - 1324)];
    else if (c < 2476) v = H2[(size_t)pi * 1024 + (c - 1452)];
    if (c < 2476) featF[(size_t)b * 2476 + c] = v;
    featB[(size_t)b * 2496 + c] = f2bf(v);
  }
}

// ---------------------------------------------------------------------------
// y[b] = o1[b,:] @ W_o2 + b_o2
// ---------------------------------------------------------------------------
__global__ void gnn_head(const float* __restrict__ o1, const float* __restrict__ W,
                         const float* __restrict__ bo, float* __restrict__ y) {
  int i = blockIdx.x * blockDim.x + threadIdx.x;
  if (i < 2048) {
    float s = bo[0];
#pragma unroll
    for (int j = 0; j < 64; ++j) s += o1[(size_t)i * 64 + j] * W[j];
    y[i] = s;
  }
}

// ---------------------------------------------------------------------------
extern "C" void kernel_launch(void* const* d_in, const int* in_sizes, int n_in,
                              void* d_out, int out_size, void* d_ws, size_t ws_size,
                              hipStream_t stream) {
  (void)in_sizes; (void)n_in; (void)out_size; (void)ws_size;

  const int*   d_index = (const int*)d_in[0];
  const int*   p_index = (const int*)d_in[1];
  const int*   p_emb   = (const int*)d_in[2];
  const float* d_vecs  = (const float*)d_in[3];
  const float* d_ecfps = (const float*)d_in[4];
  const float* d_ew    = (const float*)d_in[5];
  const float* p_gos   = (const float*)d_in[6];
  const float* p_ew    = (const float*)d_in[7];
  const float* emb     = (const float*)d_in[8];
  const float* W_ge  = (const float*)d_in[9];  const float* b_ge  = (const float*)d_in[10];
  const float* W_gg  = (const float*)d_in[11]; const float* b_gg  = (const float*)d_in[12];
  const float* g_n1  = (const float*)d_in[13]; const float* b_n1  = (const float*)d_in[14];
  const float* W_conv= (const float*)d_in[15]; const float* b_conv= (const float*)d_in[16];
  const float* W_fc  = (const float*)d_in[17]; const float* b_fc  = (const float*)d_in[18];
  const float* W_e1  = (const float*)d_in[19]; const float* b_e1  = (const float*)d_in[20];
  const float* g_e1  = (const float*)d_in[21]; const float* be_e1 = (const float*)d_in[22];
  const float* W_e2  = (const float*)d_in[23]; const float* b_e2  = (const float*)d_in[24];
  const float* g_e2  = (const float*)d_in[25]; const float* be_e2 = (const float*)d_in[26];
  const float* W_d1  = (const float*)d_in[27]; const float* b_d1  = (const float*)d_in[28];
  const float* g_d1  = (const float*)d_in[29]; const float* be_d1 = (const float*)d_in[30];
  const float* W_d2  = (const float*)d_in[31]; const float* b_d2  = (const float*)d_in[32];
  const float* g_d2  = (const float*)d_in[33]; const float* be_d2 = (const float*)d_in[34];
  const float* W_o1  = (const float*)d_in[35]; const float* b_o1  = (const float*)d_in[36];
  const float* g_o1  = (const float*)d_in[37]; const float* be_o1 = (const float*)d_in[38];
  const float* W_o2  = (const float*)d_in[39]; const float* b_o2  = (const float*)d_in[40];

  float* outY    = (float*)d_out;                       // [2048]
  float* outDec  = outY + 2048;                         // [2048,2476]
  float* outFeat = outDec + (size_t)2048 * 2476;        // [2048,2476]

  // ---- workspace bump allocator
  char* ws = (char*)d_ws;
  size_t off = 0;
  auto alloc = [&](size_t bytes) -> char* {
    char* p = ws + off;
    off += (bytes + 255) & ~(size_t)255;
    return p;
  };

  // bf16 A-side buffers (row-major, K contiguous)
  u16* ecfps_bf = (u16*)alloc((size_t)4096 * 1024 * 2);
  u16* dew_bf   = (u16*)alloc((size_t)4096 * 4096 * 2);
  u16* pgos_bf  = (u16*)alloc((size_t)4096 * 2048 * 2);
  u16* pew_bf   = (u16*)alloc((size_t)4096 * 4096 * 2);
  u16* conv_bf  = (u16*)alloc((size_t)2048 * 3872 * 2);
  u16* feat_bf  = (u16*)alloc((size_t)2048 * 2496 * 2);   // padded K
  u16* e1_bf    = (u16*)alloc((size_t)2048 * 2048 * 2);
  u16* enc_bf   = (u16*)alloc((size_t)2048 * 256 * 2);
  u16* d1_bf    = (u16*)alloc((size_t)2048 * 2048 * 2);
  // bf16 B-side buffers (pre-transposed: [N][K])
  u16* Wge_t  = (u16*)alloc((size_t)1024 * 1024 * 2);
  u16* Wgg_t  = (u16*)alloc((size_t)1024 * 2048 * 2);
  u16* G1t    = (u16*)alloc((size_t)1024 * 4096 * 2);
  u16* G2t    = (u16*)alloc((size_t)1024 * 4096 * 2);
  u16* Wfct   = (u16*)alloc((size_t)128 * 3872 * 2);
  u16* We1t   = (u16*)alloc((size_t)2048 * 2496 * 2);     // padded K
  u16* We2t   = (u16*)alloc((size_t)256 * 2048 * 2);
  u16* Wd1t   = (u16*)alloc((size_t)2048 * 256 * 2);
  u16* Wd2t   = (u16*)alloc((size_t)2476 * 2048 * 2);
  u16* Wo1t   = (u16*)alloc((size_t)64 * 256 * 2);
  // f32 buffers
  float* G1f  = (float*)alloc((size_t)4096 * 1024 * 4);
  float* H1f  = (float*)alloc((size_t)4096 * 1024 * 4);
  float* G2f  = (float*)alloc((size_t)4096 * 1024 * 4);
  float* H2f  = (float*)alloc((size_t)4096 * 1024 * 4);
  float* xtf  = (float*)alloc((size_t)2048 * 128 * 4);
  float* e1f  = (float*)alloc((size_t)2048 * 2048 * 4);
  float* encf = (float*)alloc((size_t)2048 * 256 * 4);
  float* d1f  = (float*)alloc((size_t)2048 * 2048 * 4);
  float* o1f  = (float*)alloc((size_t)2048 * 64 * 4);
  float* meanBuf = (float*)alloc(4096 * 4);
  float* varBuf  = (float*)alloc(4096 * 4);

  auto cvt = [&](const float* src, u16* dst, size_t n) {
    unsigned blocks = (unsigned)((n + 255) / 256);
    gnn_cvt_bf16<<<dim3(blocks), dim3(256), 0, stream>>>(src, dst, (unsigned long long)n);
  };
  auto cvtT = [&](const float* src, u16* dst, int R, int C, int ldDst) {
    dim3 grid((C + 31) / 32, (ldDst + 31) / 32);
    gnn_cvtT<<<grid, dim3(32, 8), 0, stream>>>(src, dst, R, C, ldDst);
  };
  auto gemm = [&](const u16* A, const u16* Bt, const float* bias, float* C,
                  int M, int N, int K) {
    dim3 grid((N + GT_BN - 1) / GT_BN, (M + GT_BM - 1) / GT_BM);
    gnn_gemm_bf16<<<grid, dim3(256), 0, stream>>>(A, Bt, bias, C, M, N, K);
  };
  auto stats = [&](const float* X, int M, int N) {
    gnn_colstats<<<dim3((N + 63) / 64), dim3(64, 4), 0, stream>>>(X, M, N, meanBuf, varBuf);
  };
  auto bn = [&](const float* X, int M, int N, const float* g, const float* b,
                float* outF, u16* outBF) {
    size_t total = (size_t)M * N;
    size_t blocks = (total + 255) / 256;
    if (blocks > 65535) blocks = 65535;
    gnn_bn_lrelu<<<dim3((unsigned)blocks), dim3(256), 0, stream>>>(X, M, N, g, b,
                                                                   meanBuf, varBuf, outF, outBF);
  };

  // ---- conversions: A-side row-major, B-side transposed [N][K]
  cvt(d_ecfps, ecfps_bf, (size_t)4096 * 1024);
  cvt(d_ew,    dew_bf,   (size_t)4096 * 4096);
  cvt(p_gos,   pgos_bf,  (size_t)4096 * 2048);
  cvt(p_ew,    pew_bf,   (size_t)4096 * 4096);
  cvtT(W_ge, Wge_t, 1024, 1024, 1024);
  cvtT(W_gg, Wgg_t, 2048, 1024, 2048);
  cvtT(W_fc, Wfct,  3872, 128,  3872);
  cvtT(W_e1, We1t,  2476, 2048, 2496);   // zero-padded K
  cvtT(W_e2, We2t,  2048, 256,  2048);
  cvtT(W_d1, Wd1t,  256,  2048, 256);
  cvtT(W_d2, Wd2t,  2048, 2476, 2048);
  cvtT(W_o1, Wo1t,  256,  64,   256);

  // ---- drug GCN branch
  gemm(ecfps_bf, Wge_t, nullptr, G1f, 4096, 1024, 1024);
  cvtT(G1f, G1t, 4096, 1024, 4096);
  gemm(dew_bf, G1t, b_ge, H1f, 4096, 1024, 4096);
  stats(H1f, 4096, 1024);
  bn(H1f, 4096, 1024, g_n1, b_n1, H1f, nullptr);

  // ---- protein GCN branch
  gemm(pgos_bf, Wgg_t, nullptr, G2f, 4096, 1024, 2048);
  cvtT(G2f, G2t, 4096, 1024, 4096);
  gemm(pew_bf, G2t, b_gg, H2f, 4096, 1024, 4096);
  stats(H2f, 4096, 1024);
  bn(H2f, 4096, 1024, g_n1, b_n1, H2f, nullptr);

  // ---- conv branch (bucketed) + FC
  gnn_conv<<<dim3(2048), dim3(256), 0, stream>>>(p_emb, W_conv, emb, b_conv, conv_bf);
  gemm(conv_bf, Wfct, b_fc, xtf, 2048, 128, 3872);

  // ---- feature assembly (f32 into d_out, bf16 mirror padded to 2496)
  gnn_feature<<<dim3(2048), dim3(256), 0, stream>>>(d_vecs, H1f, xtf, H2f,
                                                    d_index, p_index, outFeat, feat_bf);

  // ---- encoder (K padded to 2496; pad columns are zero on both operands)
  gemm(feat_bf, We1t, b_e1, e1f, 2048, 2048, 2496);
  stats(e1f, 2048, 2048);
  bn(e1f, 2048, 2048, g_e1, be_e1, e1f, e1_bf);

  gemm(e1_bf, We2t, b_e2, encf, 2048, 256, 2048);
  stats(encf, 2048, 256);
  bn(encf, 2048, 256, g_e2, be_e2, encf, enc_bf);

  // ---- decoder (dec written into d_out, BN in place)
  gemm(enc_bf, Wd1t, b_d1, d1f, 2048, 2048, 256);
  stats(d1f, 2048, 2048);
  bn(d1f, 2048, 2048, g_d1, be_d1, d1f, d1_bf);

  gemm(d1_bf, Wd2t, b_d2, outDec, 2048, 2476, 2048);
  stats(outDec, 2048, 2476);
  bn(outDec, 2048, 2476, g_d2, be_d2, outDec, nullptr);

  // ---- prediction head
  gemm(enc_bf, Wo1t, b_o1, o1f, 2048, 64, 256);
  stats(o1f, 2048, 64);
  bn(o1f, 2048, 64, g_o1, be_o1, o1f, nullptr);
  gnn_head<<<dim3(8), dim3(256), 0, stream>>>(o1f, W_o2, b_o2, outY);
}
